// CausalBilinearSelfAttention_7129645711401
// MI455X (gfx1250) — matrix operand order repaired and checked by reading, not compiled, not run on hardware
//
#include <hip/hip_runtime.h>
#include <hip/hip_bf16.h>

// ---------------------------------------------------------------------------
// CausalBilinearSelfAttention for MI455X (gfx1250), bf16 WMMA pipeline.
//   B=2, T=2048, C=1024, H=16, Dh=64
// ws layout (bf16):
//   [0*TEN) q   [1*TEN) k   [2*TEN) q2  [3*TEN) k2   (each [B][H][T][Dh])
//   [4*TEN) v   ([B][H][Dh][T]  -- transposed for contiguous B-fragments)
//   [5*TEN) z   ([B*T][1024] row-major)
//   [6*TEN) xb  (x pre-converted to bf16, [B*T][1024] row-major)
//   TEN = B*H*T*Dh = 4,194,304 elements
// ---------------------------------------------------------------------------

typedef __bf16 bf16;
typedef __attribute__((ext_vector_type(16))) __bf16 v16bf;
typedef __attribute__((ext_vector_type(8)))  __bf16 bf16x8;
typedef __attribute__((ext_vector_type(8)))  float  v8f;
typedef __attribute__((ext_vector_type(4)))  int    v4i;

typedef __attribute__((address_space(1))) v4i as1_v4i;
typedef __attribute__((address_space(3))) v4i as3_v4i;

#define BATCH   2
#define SEQ     2048
#define DMODEL  1024
#define NHEAD   16
#define HDIM    64

#if __has_builtin(__builtin_amdgcn_global_load_async_to_lds_b128)
#define HAVE_ASYNC_LDS 1
#else
#define HAVE_ASYNC_LDS 0
#endif

__device__ inline void async_copy16(const bf16* gp, bf16* lp) {
#if HAVE_ASYNC_LDS
    as1_v4i* g = (as1_v4i*)(void*)const_cast<bf16*>(gp);
    as3_v4i* l = (as3_v4i*)(void*)lp;
    __builtin_amdgcn_global_load_async_to_lds_b128(g, l, 0, 0);
#else
    *(v4i*)lp = *(const v4i*)gp;
#endif
}

__device__ inline void wait_async0() {
#if HAVE_ASYNC_LDS
#if __has_builtin(__builtin_amdgcn_s_wait_asynccnt)
    __builtin_amdgcn_s_wait_asynccnt(0);
#else
    asm volatile("s_wait_asynccnt 0" ::: "memory");
#endif
#endif
}

__device__ inline v8f wmma_bf16(v16bf a, v16bf b, v8f c) {
    return __builtin_amdgcn_wmma_f32_16x16x32_bf16(
        /*neg_a=*/false, a, /*neg_b=*/false, b,
        /*c_mod=*/(short)0, c, /*reuse_a=*/false, /*reuse_b=*/false);
}

// A-matrix 16x32 fragment: lane<16 holds K=0..15 of row
// (lane&15); lane>=16 holds K=16..31.  p points at this lane's row.
__device__ inline v16bf frag_a(const bf16* p, int lane) {
    const int oh = (lane & 16) ? 16 : 0;
    bf16x8 lo = *(const bf16x8*)(p + oh);
    bf16x8 hi = *(const bf16x8*)(p + oh + 8);
    return __builtin_shufflevector(lo, hi, 0, 1, 2, 3, 4, 5, 6, 7,
                                           8, 9, 10, 11, 12, 13, 14, 15);
}

// B-matrix 32x16 fragment: lane covers column n=(lane&15); lane<16 holds
// K=0..15, lane>=16 holds K=16..31.  p points at the K-row (contiguous in K)
// of the NT-source matrix for this lane's column.
__device__ inline v16bf frag_b(const bf16* p, int lane) {
    const int o = (lane & 16) ? 16 : 0;
    bf16x8 lo = *(const bf16x8*)(p + o);
    bf16x8 hi = *(const bf16x8*)(p + o + 8);
    return __builtin_shufflevector(lo, hi, 0, 1, 2, 3, 4, 5, 6, 7,
                                           8, 9, 10, 11, 12, 13, 14, 15);
}

__device__ inline unsigned pack2(float a, float b) {
    union { bf16 h[2]; unsigned u; } u_;
    u_.h[0] = (bf16)a; u_.h[1] = (bf16)b;
    return u_.u;
}

// ---------------------------------------------------------------------------
// x (f32) -> xb (bf16), packed pair stores.
// ---------------------------------------------------------------------------
__global__ __launch_bounds__(256) void convert_kernel(const float* __restrict__ x,
                                                      bf16* __restrict__ xb) {
    const int i = blockIdx.x * blockDim.x + threadIdx.x;   // pair index
    const float2 f = ((const float2*)x)[i];
    ((unsigned*)xb)[i] = pack2(f.x, f.y);
}

// ---------------------------------------------------------------------------
// NT GEMM:  C[m,n] = sum_k A[m,k] * Bw[n,k],  A bf16, Bw f32.
//   M=4096 N=1024 K=1024.  Block 256 thr (8 waves), tile 128x64, K-step 32.
//   A tile staged via async global->LDS b128 copies (ASYNCcnt) when available.
//   OK==0 : scatter bf16 into [B][H][T][Dh]
//   OK==1 : f32 row-major out
//   OK==2 : scatter bf16 transposed into [B][H][Dh][T]   (for V)
// ---------------------------------------------------------------------------
#define BM 128
#define BN 64
#define BK 32
#define ASTR 40   // padded LDS row stride (elements); 80B keeps 16B alignment
#define BSTR 40

template <int OK>
__global__ __launch_bounds__(256) void gemm_nt(const bf16* __restrict__ Axb,
                                               const float* __restrict__ Bw,
                                               void* __restrict__ Outp) {
    constexpr int N = DMODEL;        // 1024
    constexpr int K = DMODEL;        // 1024

    __shared__ bf16 As[BM * ASTR];
    __shared__ bf16 Bs[BN * BSTR];

    const int tid  = threadIdx.x;
    const int lane = tid & 31;
    const int wave = tid >> 5;
    const int wm   = wave & 3;   // 4 wave-rows   (32 rows each)
    const int wn   = wave >> 2;  // 2 wave-cols   (32 cols each)
    const int m0   = blockIdx.y * BM;
    const int n0   = blockIdx.x * BN;

    v8f acc[2][2] = {};

    for (int k0 = 0; k0 < K; k0 += BK) {
        // ---- stage A tile: pure bf16 copy, 16B granules (async to LDS) ----
#pragma unroll
        for (int i = 0; i < (BM * BK) / (8 * 256); ++i) {   // 2 x 16B per thread
            int idx = tid + i * 256;                        // 0..511
            int r = idx >> 2, c = (idx & 3) * 8;
            async_copy16(Axb + (size_t)(m0 + r) * K + k0 + c,
                         &As[r * ASTR + c]);
        }
        // ---- stage B tile (weights, f32 -> packed bf16 pairs) ----
#pragma unroll
        for (int i = 0; i < (BN * BK / 2) / 256; ++i) {     // 4
            int idx = tid + i * 256;
            int r = idx >> 4, c = idx & 15;
            const float2 f =
                ((const float2*)Bw)[((size_t)(n0 + r) * K + k0) / 2 + c];
            ((unsigned*)Bs)[r * (BSTR / 2) + c] = pack2(f.x, f.y);
        }
        // prefetch next K tile of weights into cache (global_prefetch_b8)
        if (k0 + BK < K)
            __builtin_prefetch(&Bw[(size_t)(n0 + (tid >> 2)) * K + k0 + BK], 0, 1);
        wait_async0();
        __syncthreads();

        v16bf af[2], bfr[2];
#pragma unroll
        for (int mi = 0; mi < 2; ++mi)
            af[mi] = frag_a(&As[(wm * 32 + mi * 16 + (lane & 15)) * ASTR], lane);
#pragma unroll
        for (int ni = 0; ni < 2; ++ni)
            bfr[ni] = frag_b(&Bs[(wn * 32 + ni * 16 + (lane & 15)) * BSTR], lane);
#pragma unroll
        for (int mi = 0; mi < 2; ++mi)
#pragma unroll
            for (int ni = 0; ni < 2; ++ni)
                acc[mi][ni] = wmma_bf16(af[mi], bfr[ni], acc[mi][ni]);
        __syncthreads();
    }

    // ---- epilogue: C layout = lane n=(lane&15), rows r + 8*(lane>=16) ----
#pragma unroll
    for (int mi = 0; mi < 2; ++mi) {
#pragma unroll
        for (int ni = 0; ni < 2; ++ni) {
            int n  = n0 + wn * 32 + ni * 16 + (lane & 15);
            int mb = m0 + wm * 32 + mi * 16 + ((lane >> 4) << 3);
#pragma unroll
            for (int r = 0; r < 8; ++r) {
                int   m   = mb + r;
                float val = acc[mi][ni][r];
                if (OK == 0) {
                    int b = m >> 11, t = m & (SEQ - 1);
                    int h = n >> 6,  d = n & (HDIM - 1);
                    ((bf16*)Outp)[((((size_t)b * NHEAD + h) * SEQ + t) << 6) + d] =
                        (bf16)val;
                } else if (OK == 2) {
                    int b = m >> 11, t = m & (SEQ - 1);
                    int h = n >> 6,  d = n & (HDIM - 1);
                    ((bf16*)Outp)[(((size_t)(b * NHEAD + h) << 6) + d) * SEQ + t] =
                        (bf16)val;
                } else {
                    ((float*)Outp)[(size_t)m * N + n] = val;
                }
            }
        }
    }
}

// ---------------------------------------------------------------------------
// RoPE over q,k,q2,k2 (4 contiguous [B][H][T][Dh] bf16 tensors in ws).
// One thread per row (t, head); 32 rotation pairs; inv_freq by recurrence.
// ---------------------------------------------------------------------------
__global__ __launch_bounds__(256) void rope_kernel(bf16* __restrict__ qk) {
    const int row = blockIdx.x * blockDim.x + threadIdx.x;  // 0 .. 4*B*H*T-1
    const int t   = row & (SEQ - 1);                        // last axis before Dh
    bf16* p = qk + (size_t)row * HDIM;
    const float tf   = (float)t;
    const float step = 0.74989420933245582f;   // 10000^(-1/32)
    float inv = 1.0f;
#pragma unroll
    for (int i = 0; i < HDIM / 2; ++i) {
        float ang = tf * inv;
        inv *= step;
        float c  = __cosf(ang);
        float s  = __sinf(ang);
        float x1 = (float)p[i];
        float x2 = (float)p[i + HDIM / 2];
        p[i]            = (bf16)(x1 * c + x2 * s);
        p[i + HDIM / 2] = (bf16)(x2 * c - x1 * s);
    }
}

// ---------------------------------------------------------------------------
// Bilinear causal attention. Wave-per-Q-tile (16 rows). Per 32-wide K chunk:
//   S = Q K^T, S2 = Q2 K2^T  (8 wmma), pattern = S*S2/4096 masked,
//   reshape via wave-private LDS, then z += pattern * V (4 wmma, contiguous
//   b128 loads thanks to the transposed V layout).
// z written bf16 row-major [B*T][1024].
// ---------------------------------------------------------------------------
__global__ __launch_bounds__(256) void attn_kernel(
    const bf16* __restrict__ Q,  const bf16* __restrict__ Kt,
    const bf16* __restrict__ Q2, const bf16* __restrict__ K2t,
    const bf16* __restrict__ VT, bf16* __restrict__ Z) {
    __shared__ bf16 patt[8][16 * 32];  // per-wave 16x32 pattern tile

    const int tid  = threadIdx.x;
    const int lane = tid & 31;
    const int wave = tid >> 5;
    const int bh   = blockIdx.y;                 // b*16 + h
    const int qi   = blockIdx.x * 8 + wave;      // 16-row q tile index
    const int q0   = qi * 16;

    const size_t base = (size_t)bh * SEQ * HDIM;
    const bf16* Qb  = Q   + base;
    const bf16* Kb  = Kt  + base;
    const bf16* Q2b = Q2  + base;
    const bf16* K2b = K2t + base;
    const bf16* VTb = VT  + base;                // [Dh][T] within this head

    // Q fragments for Dh chunks c=0 (k 0..31), c=1 (k 32..63); reused all loop
    const bf16* qrow  = Qb  + (size_t)(q0 + (lane & 15)) * HDIM;
    const bf16* q2row = Q2b + (size_t)(q0 + (lane & 15)) * HDIM;
    v16bf aq[2], aq2[2];
    aq[0]  = frag_a(qrow, lane);       aq[1]  = frag_a(qrow + 32, lane);
    aq2[0] = frag_a(q2row, lane);      aq2[1] = frag_a(q2row + 32, lane);

    v8f accz[4] = {};
    const int n  = lane & 15;
    const int mb = (lane >> 4) << 3;
    const int nkb = (qi >> 1) + 1;     // # of 32-wide K chunks (causal)

    for (int kb = 0; kb < nkb; ++kb) {
#pragma unroll
        for (int st = 0; st < 2; ++st) {
            const int kt = kb * 2 + st;          // 16-wide k tile (wave-uniform)
            bf16* pp = &patt[wave][st * 16];     // row stride 32
            if (kt <= qi) {
                const int k0 = kt * 16;
                const bf16* krow  = Kb  + (size_t)(k0 + (lane & 15)) * HDIM;
                const bf16* k2row = K2b + (size_t)(k0 + (lane & 15)) * HDIM;
                v8f s = {}, s2 = {};
                s  = wmma_bf16(aq[0],  frag_b(krow,       lane), s);
                s  = wmma_bf16(aq[1],  frag_b(krow + 32,  lane), s);
                s2 = wmma_bf16(aq2[0], frag_b(k2row,      lane), s2);
                s2 = wmma_bf16(aq2[1], frag_b(k2row + 32, lane), s2);
                const float scale = 1.0f / (float)(HDIM * HDIM);
#pragma unroll
                for (int r = 0; r < 8; ++r) {
                    float pv = s[r] * s2[r] * scale;
                    const int m = mb + r;
                    if (kt == qi && n > m) pv = 0.0f;   // causal diagonal mask
                    pp[m * 32 + n] = (bf16)pv;
                }
            } else {
#pragma unroll
                for (int r = 0; r < 8; ++r) pp[(mb + r) * 32 + n] = (bf16)0.0f;
            }
        }

        // pattern (16x32) as A fragment, straight from wave-private LDS
        v16bf pa = frag_a(&patt[wave][(lane & 15) * 32], lane);

        const int kk0 = kb * 32;
#pragma unroll
        for (int dg = 0; dg < 4; ++dg) {
            // V^T row d = dg*16+n, contiguous in t  ->  two b128 loads
            const bf16* vt = VTb + (size_t)(dg * 16 + n) * SEQ + kk0;
            accz[dg] = wmma_bf16(pa, frag_b(vt, lane), accz[dg]);
        }
    }

    // store z (bf16 row-major [B*T][1024])
    const int b = bh >> 4, h = bh & 15;
#pragma unroll
    for (int dg = 0; dg < 4; ++dg) {
#pragma unroll
        for (int r = 0; r < 8; ++r) {
            const int m = q0 + mb + r;
            Z[(size_t)(b * SEQ + m) * DMODEL + h * HDIM + dg * 16 + n] =
                (bf16)accz[dg][r];
        }
    }
}

// ---------------------------------------------------------------------------
extern "C" void kernel_launch(void* const* d_in, const int* in_sizes, int n_in,
                              void* d_out, int out_size, void* d_ws, size_t ws_size,
                              hipStream_t stream) {
    (void)in_sizes; (void)n_in; (void)out_size; (void)ws_size;

    const float* x     = (const float*)d_in[0];
    const float* Wq    = (const float*)d_in[1];
    const float* Wk    = (const float*)d_in[2];
    const float* Wq2   = (const float*)d_in[3];
    const float* Wk2   = (const float*)d_in[4];
    const float* Wv    = (const float*)d_in[5];
    const float* Wproj = (const float*)d_in[6];

    const size_t TEN = (size_t)BATCH * NHEAD * SEQ * HDIM;  // 4,194,304
    bf16* ws = (bf16*)d_ws;
    bf16* q  = ws + 0 * TEN;
    bf16* k  = ws + 1 * TEN;
    bf16* q2 = ws + 2 * TEN;
    bf16* k2 = ws + 3 * TEN;
    bf16* v  = ws + 4 * TEN;   // transposed [B][H][Dh][T]
    bf16* z  = ws + 5 * TEN;
    bf16* xb = ws + 6 * TEN;   // x as bf16

    dim3 blk(256);
    dim3 gg(DMODEL / BN, (BATCH * SEQ) / BM);  // (16, 32)

    // 0) x (f32) -> xb (bf16): read x once, feed 5 GEMMs from half-width data
    convert_kernel<<<(unsigned)(TEN / 2 / 256), blk, 0, stream>>>(x, xb);

    // 1) projections (q/k/q2/k2 head-major; v transposed)
    gemm_nt<0><<<gg, blk, 0, stream>>>(xb, Wq,  (void*)q);
    gemm_nt<0><<<gg, blk, 0, stream>>>(xb, Wk,  (void*)k);
    gemm_nt<0><<<gg, blk, 0, stream>>>(xb, Wq2, (void*)q2);
    gemm_nt<0><<<gg, blk, 0, stream>>>(xb, Wk2, (void*)k2);
    gemm_nt<2><<<gg, blk, 0, stream>>>(xb, Wv,  (void*)v);

    // 2) RoPE on q,k,q2,k2 (4*TEN elements, 64 per row)
    {
        const int rows = 4 * (int)(TEN / HDIM);  // 262,144
        rope_kernel<<<rows / 256, 256, 0, stream>>>(ws);
    }

    // 3) bilinear causal attention -> z
    attn_kernel<<<dim3(SEQ / 128, BATCH * NHEAD), blk, 0, stream>>>(q, k, q2, k2, v, z);

    // 4) output projection -> d_out (f32)
    gemm_nt<1><<<gg, blk, 0, stream>>>(z, Wproj, d_out);
}
